// CrossAttention_70617852281547
// MI455X (gfx1250) — matrix operand order repaired and checked by reading, not compiled, not run on hardware
//
#include <hip/hip_runtime.h>
#include <hip/hip_bf16.h>

typedef __attribute__((ext_vector_type(16))) __bf16 v16bf;
typedef __attribute__((ext_vector_type(8)))  float  v8f;

union FragU { uint4 q[2]; v16bf v; };

// float -> bf16 with round-to-nearest-even
__device__ __forceinline__ unsigned short f2bf(float f) {
    unsigned int u = __builtin_bit_cast(unsigned int, f);
    u += 0x7FFFu + ((u >> 16) & 1u);
    return (unsigned short)(u >> 16);
}

// 16-bit A 16x32 fragment, contiguous k order used here: lanes 0-15 hold K {0..15},
// lanes 16-31 hold K {16..31}. prow points at 32 contiguous K elems.
__device__ __forceinline__ v16bf load_fragA(const unsigned short* prow, int laneHalf) {
    FragU f;
    const unsigned short* p = prow + laneHalf * 16;
    f.q[0] = *(const uint4*)(p);
    f.q[1] = *(const uint4*)(p + 8);
    return f.v;
}

// 16-bit B 32x16 fragment, contiguous k order used here: lane holds column n=lane&15,
// K range (lane>>4)*16..+15 contiguous. p points at those 16 elems.
__device__ __forceinline__ v16bf load_fragB(const unsigned short* p) {
    FragU f;
    f.q[0] = *(const uint4*)(p);
    f.q[1] = *(const uint4*)(p + 8);
    return f.v;
}

__device__ __forceinline__ v8f wmma_bf16(v16bf a, v16bf b, v8f c) {
    return __builtin_amdgcn_wmma_f32_16x16x32_bf16(
        false, a, false, b, (short)0, c, false, false);
}

// ---- CDNA5 async global->LDS copy (16B per lane), tracked by ASYNCcnt ----
__device__ __forceinline__ void async_copy16(void* lds, const void* g) {
    // low 32 bits of a generic pointer into the shared aperture == LDS offset
    unsigned loff = (unsigned)(unsigned long long)lds;
    asm volatile("global_load_async_to_lds_b128 %0, %1, off"
                 :: "v"(loff), "v"(g) : "memory");
}
__device__ __forceinline__ void wait_async0() {
    asm volatile("s_wait_asynccnt 0x0" ::: "memory");
}

// ---------------------------------------------------------------- convert
__global__ void cvt_bf16_kernel(const float* __restrict__ in,
                                unsigned short* __restrict__ out, int n) {
    int i = (blockIdx.x * blockDim.x + threadIdx.x) * 4;
    if (i + 3 < n) {
        float4 f = *(const float4*)(in + i);
        uint2 p;
        p.x = (unsigned)f2bf(f.x) | ((unsigned)f2bf(f.y) << 16);
        p.y = (unsigned)f2bf(f.z) | ((unsigned)f2bf(f.w) << 16);
        *(uint2*)(out + i) = p;
    } else {
        for (int j = 0; j < 4 && i + j < n; ++j) out[i + j] = f2bf(in[i + j]);
    }
}

// ---------------------------------------------------------------- GEMM
// C[M,N] = A[M,K] * W[K,N], bf16 in, f32 out. Tile 128x128, Kstep 32,
// double-buffered LDS with async A staging. 256 threads = 8 waves;
// wave w: rows (w>>1)*32 (2 row-tiles), cols (w&1)*64 (4 col-tiles) -> 8 WMMA.
__global__ __launch_bounds__(256) void gemm_bf16_kernel(
    const unsigned short* __restrict__ A, const unsigned short* __restrict__ W,
    float* __restrict__ C, int M, int N, int K) {
    __shared__ unsigned short sA[2][128 * 32];   // [m][k]
    __shared__ unsigned short sBt[2][128 * 32];  // transposed: [n][k]
    const int t = threadIdx.x;
    const int lane = t & 31, wv = t >> 5;
    const int m0 = blockIdx.y * 128, n0 = blockIdx.x * 128;
    const int wr = wv >> 1, wc = wv & 1;
    const int fr = lane & 15, half = lane >> 4, kb = half * 16;
    // A staging: 128x32, 2x async b128 per thread
    const int arow = t >> 2, acol = (t & 3) * 8;
    // W staging: 32x128 -> transposed, 16 elems per thread
    const int krow = t >> 3, nseg = (t & 7) * 16;

    v8f acc[2][4] = {};

    auto stage = [&](int buf, int k0) {
        // async A (contiguous both sides)
        async_copy16(&sA[buf][arow * 32 + acol],
                     A + (size_t)(m0 + arow) * K + k0 + acol);
        async_copy16(&sA[buf][(arow + 64) * 32 + acol],
                     A + (size_t)(m0 + arow + 64) * K + k0 + acol);
        // W transposed (load + scatter ds_store_b16)
        const uint4* wsrc = (const uint4*)(W + (size_t)(k0 + krow) * N + n0 + nseg);
        uint4 w0 = wsrc[0], w1 = wsrc[1];
        const unsigned short* we = (const unsigned short*)&w0;
#pragma unroll
        for (int j = 0; j < 8; ++j) sBt[buf][(nseg + j) * 32 + krow] = we[j];
        we = (const unsigned short*)&w1;
#pragma unroll
        for (int j = 0; j < 8; ++j) sBt[buf][(nseg + 8 + j) * 32 + krow] = we[j];
    };

    stage(0, 0);
    wait_async0();
    __syncthreads();

    int buf = 0;
    for (int k0 = 0; k0 < K; k0 += 32) {
        const int nk = k0 + 32;
        if (nk < K) stage(buf ^ 1, nk);  // overlap next stage with compute

        const unsigned short* a0p = &sA[buf][(wr * 32 + fr) * 32];
        v16bf a0 = load_fragA(a0p, half);
        v16bf a1 = load_fragA(a0p + 16 * 32, half);
#pragma unroll
        for (int ct = 0; ct < 4; ++ct) {
            v16bf b = load_fragB(&sBt[buf][(wc * 64 + ct * 16 + fr) * 32 + kb]);
            acc[0][ct] = wmma_bf16(a0, b, acc[0][ct]);
            acc[1][ct] = wmma_bf16(a1, b, acc[1][ct]);
        }
        if (nk < K) wait_async0();
        __syncthreads();
        buf ^= 1;
    }

#pragma unroll
    for (int rt = 0; rt < 2; ++rt) {
        const int rowBase = m0 + wr * 32 + rt * 16 + 8 * half;
#pragma unroll
        for (int ct = 0; ct < 4; ++ct)
#pragma unroll
            for (int i = 0; i < 8; ++i)
                C[(size_t)(rowBase + i) * N + n0 + wc * 64 + ct * 16 + fr] =
                    acc[rt][ct][i];
    }
}

// ---------------------------------------------------------------- LayerNorm
// One wave per (b,s,h); hd=64 -> 2 elems/lane. Output bf16 head-major [B,H,S,64].
__global__ __launch_bounds__(128) void ln_head_kernel(
    const float* __restrict__ X, const float* __restrict__ g,
    const float* __restrict__ bta, unsigned short* __restrict__ Y,
    int Bn, int S, int H, int Dm, int applyLn) {
    int wid = (blockIdx.x * blockDim.x + threadIdx.x) >> 5;
    int lane = threadIdx.x & 31;
    int total = Bn * S * H;
    if (wid >= total) return;
    int h = wid % H;
    int s = (wid / H) % S;
    int b = wid / (H * S);
    const float* x = X + ((size_t)b * S + s) * Dm + h * 64;
    float2 xv = *(const float2*)(x + lane * 2);
    float y0 = xv.x, y1 = xv.y;
    if (applyLn) {
        float sum = xv.x + xv.y;
#pragma unroll
        for (int m = 1; m < 32; m <<= 1) sum += __shfl_xor(sum, m, 32);
        float mu = sum * (1.0f / 64.0f);
        float d0 = xv.x - mu, d1 = xv.y - mu;
        float ss = d0 * d0 + d1 * d1;
#pragma unroll
        for (int m = 1; m < 32; m <<= 1) ss += __shfl_xor(ss, m, 32);
        float inv = rsqrtf(ss * (1.0f / 64.0f) + 1e-5f);
        y0 = d0 * inv * g[lane * 2]     + bta[lane * 2];
        y1 = d1 * inv * g[lane * 2 + 1] + bta[lane * 2 + 1];
    }
    unsigned int packed = (unsigned)f2bf(y0) | ((unsigned)f2bf(y1) << 16);
    unsigned int* out32 =
        (unsigned int*)(Y + (((size_t)b * H + h) * S + s) * 64);
    out32[lane] = packed;
}

// ---------------------------------------------------------------- attention
// Flash attention. grid(Sq/64, B*H), 128 threads = 4 waves; wave owns 16 q-rows.
__global__ __launch_bounds__(128) void attn_kernel(
    const unsigned short* __restrict__ Qn, const unsigned short* __restrict__ Kn,
    const unsigned short* __restrict__ Vn, unsigned short* __restrict__ Oa,
    int Bn, int H, int Sq, int Skv, int Dm) {
    __shared__ unsigned short sQ[64 * 64];     // [q][d]
    __shared__ unsigned short sK[32 * 64];     // [kv][d]
    __shared__ unsigned short sVT[64 * 32];    // transposed: [d][kv]
    __shared__ unsigned short sP[4 * 16 * 32]; // per-wave P tile [q][kv]
    const int t = threadIdx.x, lane = t & 31, wv = t >> 5;
    const int bh = blockIdx.y;
    const int b = bh / H, h = bh % H;
    const int q0 = blockIdx.x * 64;
    const unsigned short* Qbase = Qn + ((size_t)bh * Sq + q0) * 64;
    const unsigned short* Kbase = Kn + (size_t)bh * Skv * 64;
    const unsigned short* Vbase = Vn + (size_t)bh * Skv * 64;

    {   // stage Q tile 64x64 via async copies (32 elems = 4x b128 per thread)
        int row = t >> 1, hc = (t & 1) * 32;
#pragma unroll
        for (int j = 0; j < 4; ++j)
            async_copy16(&sQ[row * 64 + hc + j * 8],
                         Qbase + (size_t)row * 64 + hc + j * 8);
    }
    wait_async0();
    __syncthreads();

    const int fr = lane & 15, half = lane >> 4, kb = half * 16;
    v16bf aQ0 = load_fragA(&sQ[(wv * 16 + fr) * 64 + 0],  half);
    v16bf aQ1 = load_fragA(&sQ[(wv * 16 + fr) * 64 + 32], half);
    v8f acc[4] = {};
    float mrow[8], lrow[8];
#pragma unroll
    for (int i = 0; i < 8; ++i) { mrow[i] = -1e30f; lrow[i] = 0.0f; }

    for (int kv0 = 0; kv0 < Skv; kv0 += 32) {
        {   // stage K row-major via async; V transposed manually
            int r = t >> 2, qtr = (t & 3) * 16;
            async_copy16(&sK[r * 64 + qtr],
                         Kbase + (size_t)(kv0 + r) * 64 + qtr);
            async_copy16(&sK[r * 64 + qtr + 8],
                         Kbase + (size_t)(kv0 + r) * 64 + qtr + 8);
            const uint4* vsrc = (const uint4*)(Vbase + (size_t)(kv0 + r) * 64 + qtr);
            uint4 v0 = vsrc[0], v1 = vsrc[1];
            const unsigned short* ve = (const unsigned short*)&v0;
#pragma unroll
            for (int j = 0; j < 8; ++j) sVT[(qtr + j) * 32 + r] = ve[j];
            ve = (const unsigned short*)&v1;
#pragma unroll
            for (int j = 0; j < 8; ++j) sVT[(qtr + 8 + j) * 32 + r] = ve[j];
        }
        if (kv0 + 32 < Skv) {   // global_prefetch of next V chunk
            __builtin_prefetch(Vbase + (size_t)(kv0 + 32) * 64 + t * 16, 0, 0);
        }
        wait_async0();
        __syncthreads();

        // S = Q * K^T : wave computes its 16x32 slice as two 16x16 tiles
        v8f s0 = {}, s1 = {};
        v16bf b00 = load_fragB(&sK[fr * 64 + 0  + kb]);
        v16bf b01 = load_fragB(&sK[fr * 64 + 32 + kb]);
        v16bf b10 = load_fragB(&sK[(16 + fr) * 64 + 0  + kb]);
        v16bf b11 = load_fragB(&sK[(16 + fr) * 64 + 32 + kb]);
        s0 = wmma_bf16(aQ0, b00, s0);
        s0 = wmma_bf16(aQ1, b01, s0);
        s1 = wmma_bf16(aQ0, b10, s1);
        s1 = wmma_bf16(aQ1, b11, s1);
        s0 = s0 * 0.125f;   // 1/sqrt(64)
        s1 = s1 * 0.125f;

        // online softmax update (rows i+8*half; cols spread over 16 lanes/half)
        unsigned short* myP = &sP[wv * 512];
#pragma unroll
        for (int i = 0; i < 8; ++i) {
            float a0 = s0[i], a1 = s1[i];
            float mx = fmaxf(a0, a1);
#pragma unroll
            for (int msk = 1; msk < 16; msk <<= 1)
                mx = fmaxf(mx, __shfl_xor(mx, msk, 32));
            float nm = fmaxf(mrow[i], mx);
            float corr = __expf(mrow[i] - nm);
            float p0 = __expf(a0 - nm), p1 = __expf(a1 - nm);
            float ps = p0 + p1;
#pragma unroll
            for (int msk = 1; msk < 16; msk <<= 1)
                ps += __shfl_xor(ps, msk, 32);
            lrow[i] = lrow[i] * corr + ps;
            mrow[i] = nm;
            acc[0][i] *= corr; acc[1][i] *= corr;
            acc[2][i] *= corr; acc[3][i] *= corr;
            int r = i + 8 * half;
            myP[r * 32 + fr]      = f2bf(p0);
            myP[r * 32 + 16 + fr] = f2bf(p1);
        }

        // O += P * V  (P via LDS C->A layout redistribution; V transposed)
        v16bf aP = load_fragA(&myP[fr * 32], half);
#pragma unroll
        for (int dg = 0; dg < 4; ++dg) {
            v16bf bV = load_fragB(&sVT[(dg * 16 + fr) * 32 + kb]);
            acc[dg] = wmma_bf16(aP, bV, acc[dg]);
        }
        __syncthreads();
    }

    // epilogue: normalize, de-transpose heads, write bf16 [B,Sq,D]
#pragma unroll
    for (int i = 0; i < 8; ++i) {
        float inv = 1.0f / lrow[i];
        size_t rowOff = ((size_t)b * Sq + q0 + wv * 16 + i + 8 * half) * (size_t)Dm + h * 64;
#pragma unroll
        for (int dg = 0; dg < 4; ++dg)
            Oa[rowOff + dg * 16 + fr] = f2bf(acc[dg][i] * inv);
    }
}

// ---------------------------------------------------------------- host
extern "C" void kernel_launch(void* const* d_in, const int* in_sizes, int n_in,
                              void* d_out, int out_size, void* d_ws, size_t ws_size,
                              hipStream_t stream) {
    (void)in_sizes; (void)n_in; (void)out_size; (void)ws_size;
    constexpr int B = 2, Sq = 4096, Skv = 1024, D = 1024, H = 16;

    const float* hid = (const float*)d_in[0];
    const float* enc = (const float*)d_in[1];
    const float* Wq  = (const float*)d_in[2];
    const float* Wk  = (const float*)d_in[3];
    const float* Wv  = (const float*)d_in[4];
    const float* Wo  = (const float*)d_in[5];
    const float* gq  = (const float*)d_in[6];
    const float* bq  = (const float*)d_in[7];
    const float* gk  = (const float*)d_in[8];
    const float* bk  = (const float*)d_in[9];

    char* ws = (char*)d_ws;
    size_t off = 0;
    auto take = [&](size_t bytes) -> void* {
        void* p = ws + off;
        off += (bytes + 255) & ~(size_t)255;
        return p;
    };
    unsigned short* hbf = (unsigned short*)take((size_t)B * Sq * D * 2);
    unsigned short* ebf = (unsigned short*)take((size_t)B * Skv * D * 2);
    unsigned short* wqb = (unsigned short*)take((size_t)D * D * 2);
    unsigned short* wkb = (unsigned short*)take((size_t)D * D * 2);
    unsigned short* wvb = (unsigned short*)take((size_t)D * D * 2);
    unsigned short* wob = (unsigned short*)take((size_t)D * D * 2);
    float* Qf = (float*)take((size_t)B * Sq * D * 4);
    float* Kf = (float*)take((size_t)B * Skv * D * 4);
    float* Vf = (float*)take((size_t)B * Skv * D * 4);
    unsigned short* Qh = (unsigned short*)take((size_t)B * Sq * D * 2);
    unsigned short* Kh = (unsigned short*)take((size_t)B * Skv * D * 2);
    unsigned short* Vh = (unsigned short*)take((size_t)B * Skv * D * 2);
    unsigned short* Oa = (unsigned short*)Qf;  // reuse dead Qf32 region

    auto cvt = [&](const float* src, unsigned short* dst, int n) {
        cvt_bf16_kernel<<<(n + 1023) / 1024, 256, 0, stream>>>(src, dst, n);
    };
    cvt(hid, hbf, B * Sq * D);
    cvt(enc, ebf, B * Skv * D);
    cvt(Wq, wqb, D * D);
    cvt(Wk, wkb, D * D);
    cvt(Wv, wvb, D * D);
    cvt(Wo, wob, D * D);

    gemm_bf16_kernel<<<dim3(D / 128, (B * Sq) / 128), 256, 0, stream>>>(
        hbf, wqb, Qf, B * Sq, D, D);
    gemm_bf16_kernel<<<dim3(D / 128, (B * Skv) / 128), 256, 0, stream>>>(
        ebf, wkb, Kf, B * Skv, D, D);
    gemm_bf16_kernel<<<dim3(D / 128, (B * Skv) / 128), 256, 0, stream>>>(
        ebf, wvb, Vf, B * Skv, D, D);

    int qWaves = B * Sq * H;
    int kWaves = B * Skv * H;
    ln_head_kernel<<<qWaves / 4, 128, 0, stream>>>(Qf, gq, bq, Qh, B, Sq, H, D, 1);
    ln_head_kernel<<<kWaves / 4, 128, 0, stream>>>(Kf, gk, bk, Kh, B, Skv, H, D, 1);
    ln_head_kernel<<<kWaves / 4, 128, 0, stream>>>(Vf, gk, bk, Vh, B, Skv, H, D, 0);

    attn_kernel<<<dim3(Sq / 64, B * H), 128, 0, stream>>>(
        Qh, Kh, Vh, Oa, B, H, Sq, Skv, D);

    gemm_bf16_kernel<<<dim3(D / 128, (B * Sq) / 128), 256, 0, stream>>>(
        Oa, wob, (float*)d_out, B * Sq, D, D);
}
